// MoELayer_1013612282518
// MI455X (gfx1250) — compile-verified, matrix operand order checked by reading, not run on hardware
//
#include <hip/hip_runtime.h>
#include <hip/hip_bf16.h>

#define E_EXP 8
#define DM    1024
#define DF    4096
#define NTOK  8192   // B*S = 4*2048
#define MT    32     // tokens per workgroup tile
#define QF    1024   // columns of F processed per quarter
#define NTHR  512    // 16 waves

typedef __attribute__((ext_vector_type(16))) __bf16 v16bf;
typedef __attribute__((ext_vector_type(8)))  float  v8f;

// ---- Workspace layout (bytes) ----
//   [0, 256)            : per-expert token counts (8 ints)
//   [1024, +256K)       : per-expert token lists
//   [W1T_OFF, +64MB)    : w1 transposed to [E][F][D] bf16 (K-contiguous rows)
//   [W2T_OFF, +64MB)    : w2 transposed to [E][D][F] bf16 (K-contiguous rows)
#define WS_LISTS_OFF 1024
#define WS_W1T_OFF   (WS_LISTS_OFF + (size_t)E_EXP * NTOK * 4)
#define WS_W2T_OFF   (WS_W1T_OFF + (size_t)E_EXP * DM * DF * 2)

// Opaque 32-bit value: defeats LICM (no hoisting LDS fragment loads out of
// outer loops into scratch spills); alignment is asserted by the uint4
// access type below, so b128 merging survives the opacity.
__device__ __forceinline__ int opaque_i(int v) {
  asm volatile("" : "+v"(v));
  return v;
}

// 16x32 bf16 WMMA fragment (A layout; B symmetric) from a row-major,
// K-contiguous buffer (LDS or global -- addrspace inference picks
// ds_load_b128 vs global_load_b128).  p = row base + kbase; kh = lane>>4.
// VGPR v<4 -> K = kh*8 + 2v ; v>=4 -> K = 16 + kh*8 + 2(v-4).
// uint4 accesses carry 16B alignment by type => always b128.
__device__ __forceinline__ v16bf frag16(const __bf16* p, int kh) {
  union { v16bf v; uint4 q[2]; } f;
  const __bf16* q = p + kh * 8;
  f.q[0] = *(const uint4*)(q);        // K offsets {0,2,4,6}+kh*8
  f.q[1] = *(const uint4*)(q + 16);   // K offsets {16,18,20,22}+kh*8
  return f.v;
}

__global__ void zero_counts_kernel(int* counts) {
  if (threadIdx.x < E_EXP) counts[threadIdx.x] = 0;
}

// One wave (32 lanes) per token: gate logits -> argmax -> append to expert bucket.
__global__ __launch_bounds__(256) void gate_topk_kernel(
    const float* __restrict__ x, const float* __restrict__ gw,
    const float* __restrict__ gb, int* __restrict__ counts,
    int* __restrict__ lists) {
  int gt   = blockIdx.x * blockDim.x + threadIdx.x;
  int tok  = gt >> 5;
  int lane = gt & 31;
  if (tok >= NTOK) return;
  const float* xr = x + (size_t)tok * DM;
  float acc[E_EXP];
#pragma unroll
  for (int e = 0; e < E_EXP; e++) acc[e] = 0.f;
  for (int d = lane; d < DM; d += 32) {
    float xv = xr[d];
    const float* g = gw + (size_t)d * E_EXP;
#pragma unroll
    for (int e = 0; e < E_EXP; e++) acc[e] += xv * g[e];
  }
#pragma unroll
  for (int e = 0; e < E_EXP; e++) {
#pragma unroll
    for (int off = 16; off > 0; off >>= 1) acc[e] += __shfl_xor(acc[e], off, 32);
  }
  if (lane == 0) {
    int best = 0;
    float bv = acc[0] + gb[0];
#pragma unroll
    for (int e = 1; e < E_EXP; e++) {
      float v = acc[e] + gb[e];
      if (v > bv) { bv = v; best = e; }  // strict > : first-max, like jnp.argmax
    }
    int pos = atomicAdd(&counts[best], 1);
    lists[best * NTOK + pos] = tok;
  }
}

// Transpose+convert one expert slab: src [R][C] fp32 -> dst [C][R] bf16.
__global__ __launch_bounds__(256) void transpose_cvt_kernel(
    const float* __restrict__ src, __bf16* __restrict__ dst, int R, int C) {
  __shared__ float tile[64][65];
  const size_t eoff = (size_t)blockIdx.z * R * C;
  const float* s = src + eoff;
  __bf16*      d = dst + eoff;
  const int r0 = blockIdx.y * 64;
  const int c0 = blockIdx.x * 64;
  for (int i = threadIdx.x; i < 64 * 64; i += 256) {
    int r = i >> 6, c = i & 63;                    // coalesced along C
    tile[r][c] = s[(size_t)(r0 + r) * C + c0 + c];
  }
  __syncthreads();
  for (int i = threadIdx.x; i < 64 * 64; i += 256) {
    int c = i >> 6, r = i & 63;                    // coalesced along R
    d[(size_t)(c0 + c) * R + r0 + r] = (__bf16)tile[r][c];
  }
}

// Fused expert FFN on a gathered 32-token tile, bf16 WMMA / f32 accumulate.
// 16 waves; each owns a 16-col strip of 256-wide chunks.  B fragments stream
// straight from pre-transposed global weights (2x global_load_b128 per lane),
// A fragments from LDS (2x ds_load_b128 per tile).
__global__ __launch_bounds__(NTHR) void moe_ffn_kernel(
    const float*  __restrict__ x,   const __bf16* __restrict__ w1t,
    const float*  __restrict__ b1,  const __bf16* __restrict__ w2t,
    const float*  __restrict__ b2,  const int* __restrict__ counts,
    const int*    __restrict__ lists, float* __restrict__ out) {
  __shared__ __bf16 Xs[MT][DM + 8];   // 32 x 1032 bf16 = 64.5 KB (16B-aligned rows)
  __shared__ __bf16 Hs[MT][QF + 8];   // 32 x 1032 bf16 = 64.5 KB
  __shared__ int    toks[MT];

  const int e    = blockIdx.y;
  const int cnt  = counts[e];
  const int base = blockIdx.x * MT;
  if (base >= cnt) return;            // uniform per block

  const int tid  = threadIdx.x;
  const int lane = tid & 31;
  const int wv   = tid >> 5;          // wave 0..15
  const int nl   = lane & 15;
  const int kh   = lane >> 4;

  if (tid < MT) {
    int idx = base + tid;
    toks[tid] = (idx < cnt) ? lists[e * NTOK + idx] : -1;
  }
  __syncthreads();

  // Gather + convert X tile into LDS (coalesced along D).
  for (int i = tid; i < MT * DM; i += NTHR) {
    int m = i >> 10;
    int k = i & (DM - 1);
    int t = toks[m];
    float v = (t >= 0) ? x[(size_t)t * DM + k] : 0.f;
    Xs[m][k] = (__bf16)v;
  }
  __syncthreads();

  const __bf16* W1e = w1t + (size_t)e * DM * DF;  // [F][D] bf16
  const __bf16* W2e = w2t + (size_t)e * DF * DM;  // [D][F] bf16

  const v8f vzero = {0.f, 0.f, 0.f, 0.f, 0.f, 0.f, 0.f, 0.f};
  v8f acc[4][2];                       // wave wv owns out cols {j*256 + wv*16 .. +16}
#pragma unroll
  for (int j = 0; j < 4; j++) { acc[j][0] = vzero; acc[j][1] = vzero; }

  for (int q = 0; q < 4; q++) {
    // ---- pass 1: Hs = relu(X @ W1[:, q*QF .. +QF] + b1) ----
    for (int c = 0; c < 4; c++) {                  // 4 chunks of 256 cols
      const int fc = q * QF + c * 256;             // this wave: cols fc+wv*16 ..+16
      const __bf16* brow = W1e + (size_t)(fc + wv * 16 + nl) * DM;
      v8f h0 = vzero, h1 = vzero;
#pragma unroll 4
      for (int kp = 0; kp < DM; kp += 32) {
        v16bf bfrag = frag16(brow + kp, kh);
        __builtin_prefetch(brow + kp + 256, 0, 3);   // speculative: OOB is dropped
        const int kx = opaque_i(kp);
        v16bf a0 = frag16(&Xs[nl][kx], kh);
        v16bf a1 = frag16(&Xs[16 + nl][kx], kh);
        h0 = __builtin_amdgcn_wmma_f32_16x16x32_bf16(false, a0, false, bfrag,
                                                     (short)0, h0, false, false);
        h1 = __builtin_amdgcn_wmma_f32_16x16x32_bf16(false, a1, false, bfrag,
                                                     (short)0, h1, false, false);
      }
      const int   lc  = c * 256 + wv * 16 + nl;
      const float bia = b1[(size_t)e * DF + fc + wv * 16 + nl];
#pragma unroll
      for (int r = 0; r < 8; r++) {
        float v0 = h0[r] + bia;
        float v1 = h1[r] + bia;
        Hs[kh * 8 + r][lc]      = (__bf16)(v0 > 0.f ? v0 : 0.f);
        Hs[16 + kh * 8 + r][lc] = (__bf16)(v1 > 0.f ? v1 : 0.f);
      }
    }
    __syncthreads();   // Hs fully written before pass 2 reads
    // ---- pass 2 (partial): acc += H_q @ W2[q*QF .. +QF, :] ----
    for (int j = 0; j < 4; j++) {
      const __bf16* brow = W2e + (size_t)(j * 256 + wv * 16 + nl) * DF + q * QF;
#pragma unroll 4
      for (int kp = 0; kp < QF; kp += 32) {
        v16bf bfrag = frag16(brow + kp, kh);
        __builtin_prefetch(brow + kp + 256, 0, 3);   // speculative: OOB is dropped
        const int kx = opaque_i(kp);
        v16bf a0 = frag16(&Hs[nl][kx], kh);
        v16bf a1 = frag16(&Hs[16 + nl][kx], kh);
        acc[j][0] = __builtin_amdgcn_wmma_f32_16x16x32_bf16(false, a0, false, bfrag,
                                                            (short)0, acc[j][0], false, false);
        acc[j][1] = __builtin_amdgcn_wmma_f32_16x16x32_bf16(false, a1, false, bfrag,
                                                            (short)0, acc[j][1], false, false);
      }
    }
    __syncthreads();   // pass 2 reads done before next quarter overwrites Hs
  }

  // ---- epilogue: bias + scatter rows back to tokens ----
#pragma unroll
  for (int j = 0; j < 4; j++) {
    const int   col = j * 256 + wv * 16 + nl;
    const float bia = b2[(size_t)e * DM + col];
#pragma unroll
    for (int mi = 0; mi < 2; mi++) {
#pragma unroll
      for (int r = 0; r < 8; r++) {
        int m = mi * 16 + kh * 8 + r;   // C-layout: lanes 0-15 -> M 0..7, 16-31 -> M 8..15
        int t = toks[m];
        if (t >= 0) out[(size_t)t * DM + col] = acc[j][mi][r] + bia;
      }
    }
  }
}

extern "C" void kernel_launch(void* const* d_in, const int* in_sizes, int n_in,
                              void* d_out, int out_size, void* d_ws, size_t ws_size,
                              hipStream_t stream) {
  (void)in_sizes; (void)n_in; (void)out_size; (void)ws_size;
  const float* x  = (const float*)d_in[0];
  const float* w1 = (const float*)d_in[1];
  const float* b1 = (const float*)d_in[2];
  const float* w2 = (const float*)d_in[3];
  const float* b2 = (const float*)d_in[4];
  const float* gw = (const float*)d_in[5];
  const float* gb = (const float*)d_in[6];
  float* out = (float*)d_out;

  int*    counts = (int*)d_ws;
  int*    lists  = (int*)((char*)d_ws + WS_LISTS_OFF);
  __bf16* w1t    = (__bf16*)((char*)d_ws + WS_W1T_OFF);   // [E][F][D]
  __bf16* w2t    = (__bf16*)((char*)d_ws + WS_W2T_OFF);   // [E][D][F]

  zero_counts_kernel<<<1, 32, 0, stream>>>(counts);
  gate_topk_kernel<<<(NTOK * 32) / 256, 256, 0, stream>>>(x, gw, gb, counts, lists);

  // One-time per launch: weights fp32 -> bf16, K-contiguous transposed layout.
  dim3 g1(DF / 64, DM / 64, E_EXP);   // w1 [D][F] -> [F][D]
  transpose_cvt_kernel<<<g1, 256, 0, stream>>>(w1, w1t, DM, DF);
  dim3 g2(DM / 64, DF / 64, E_EXP);   // w2 [F][D] -> [D][F]
  transpose_cvt_kernel<<<g2, 256, 0, stream>>>(w2, w2t, DF, DM);

  dim3 grid(NTOK / MT, E_EXP);        // tile-major: same-expert WGs adjacent (L2 reuse)
  moe_ffn_kernel<<<grid, NTHR, 0, stream>>>(x, w1t, b1, w2t, b2, counts, lists, out);
}
